// Attention_17179869253
// MI455X (gfx1250) — matrix operand order repaired and checked by reading, not compiled, not run on hardware
//
#include <hip/hip_runtime.h>
#include <math.h>

// ---------------------------------------------------------------------------
// CDNA5 (gfx1250) types & helpers
// ---------------------------------------------------------------------------
typedef __attribute__((ext_vector_type(16))) __bf16 v16bf;
typedef __attribute__((ext_vector_type(16))) unsigned short v16us;
typedef __attribute__((ext_vector_type(8)))  float v8f;

typedef __attribute__((ext_vector_type(4))) unsigned tdm_v4u;
typedef __attribute__((ext_vector_type(8))) int      tdm_v8i;
typedef __attribute__((ext_vector_type(4))) int      tdm_v4i;

union Frag { uint4 q[2]; v16us v; };

__device__ __forceinline__ unsigned short f2bf(float f) {
  unsigned u = __builtin_bit_cast(unsigned, f);
  unsigned r = u + 0x7FFFu + ((u >> 16) & 1u);   // round-to-nearest-even
  return (unsigned short)(r >> 16);
}
__device__ __forceinline__ float bf2f(unsigned short h) {
  return __builtin_bit_cast(float, (unsigned)h << 16);
}

__device__ __forceinline__ v8f wmma_bf(const Frag& a, const Frag& b, v8f c) {
  return __builtin_amdgcn_wmma_f32_16x16x32_bf16(
      false, __builtin_bit_cast(v16bf, a.v),
      false, __builtin_bit_cast(v16bf, b.v),
      (short)0, c, false, false);
}

// ---------------------------------------------------------------------------
// Tensor Data Mover: async 2D bf16 tile -> LDS (with per-row LDS padding).
// D# packed per CDNA5 ISA.
//   pad_interval_code: 0=2,1=4,2=8,3=16,4=32,...  DWORDs between pads
//   pad_amount_code:   pad DWORDs - 1
// ---------------------------------------------------------------------------
#if __has_include(<hip/amd_detail/amd_gfx1250_TDM.h>)
#define TDM_6ARG 1
#endif

__device__ __forceinline__ void tdm_load_2d_bf16(
    unsigned lds_addr, const void* gaddr,
    unsigned tensor_d0, unsigned tensor_d1, unsigned d0_stride,
    unsigned tile_d0, unsigned tile_d1,
    unsigned pad_interval_code, unsigned pad_amount_code) {
  unsigned long long ga = (unsigned long long)gaddr;
  tdm_v4u g0;
  g0[0] = 1u;                                        // count=1, user mode
  g0[1] = lds_addr;                                  // LDS byte address
  g0[2] = (unsigned)(ga & 0xffffffffu);              // global addr lo
  g0[3] = (unsigned)((ga >> 32) & 0x01ffffffu) | (2u << 30);  // hi | type=2
  unsigned w0 = (1u << 16)                           // data_size = 2 bytes
              | (1u << 20)                           // pad_enable
              | (pad_interval_code << 22)
              | (pad_amount_code << 25);
  tdm_v8i g1;
  g1[0] = (int)w0;
  g1[1] = (int)((tensor_d0 & 0xffffu) << 16);                         // dim0 lo16
  g1[2] = (int)(((tensor_d0 >> 16) & 0xffffu) | ((tensor_d1 & 0xffffu) << 16));
  g1[3] = (int)(((tensor_d1 >> 16) & 0xffffu) | ((tile_d0 & 0xffffu) << 16));
  g1[4] = (int)(tile_d1 & 0xffffu);                                   // tile_dim1, tile_dim2=0
  g1[5] = (int)d0_stride;                                             // dim0_stride lo32
  g1[6] = 0;                                                          // dim0_stride hi | dim1_stride lo
  g1[7] = 0;
  tdm_v4i z4 = {0, 0, 0, 0};
#ifdef TDM_6ARG
  tdm_v8i z8 = {0, 0, 0, 0, 0, 0, 0, 0};
  __builtin_amdgcn_tensor_load_to_lds(g0, g1, z4, z4, z8, 0);
#else
  __builtin_amdgcn_tensor_load_to_lds(g0, g1, z4, z4, 0);
#endif
}

__device__ __forceinline__ unsigned lds_addr_of(const void* p) {
  return (unsigned)(unsigned long long)p;   // low 32 bits = LDS offset
}

// ---------------------------------------------------------------------------
// Problem constants
// ---------------------------------------------------------------------------
#define BATCH 8
#define HH 32
#define WW 32
#define CC 768
#define NH 12
#define HD 64
#define HW 1024          // HH*WW
#define BH 96            // BATCH*NH
#define ROWS 8192        // BATCH*HW

// ---------------------------------------------------------------------------
// fp32 -> bf16 convert (vector of 4)
// ---------------------------------------------------------------------------
__global__ __launch_bounds__(256) void cvt_bf16(const float* __restrict__ in,
                                                unsigned short* __restrict__ out,
                                                int n) {
  int i = (blockIdx.x * 256 + threadIdx.x) * 4;
  if (i >= n) return;
  float4 f = *(const float4*)(in + i);
  uint2 o;
  o.x = (unsigned)f2bf(f.x) | ((unsigned)f2bf(f.y) << 16);
  o.y = (unsigned)f2bf(f.z) | ((unsigned)f2bf(f.w) << 16);
  *(uint2*)(out + i) = o;
}

// ---------------------------------------------------------------------------
// Tiled bf16 WMMA GEMM: C(128x128 tile) = A(MxK) @ B(KxN) + bias
//  A tile staged by TDM (async tensor -> LDS); B tile manual transpose.
//  mode 0: QKV epilogue -> scatter into per-head q/k/v bf16 (k pre-scaled 0.125)
//  mode 1: plain fp32 store to Out
// ---------------------------------------------------------------------------
__global__ __launch_bounds__(256) void gemm_bf16(
    const unsigned short* __restrict__ A,   // (M,K) bf16 row-major
    const unsigned short* __restrict__ B,   // (K,N) bf16 row-major
    const float* __restrict__ bias,         // (N)
    int K, int N, int mode,
    float* __restrict__ Out,
    unsigned short* __restrict__ qb,
    unsigned short* __restrict__ kb,
    unsigned short* __restrict__ vb) {
  __shared__ __align__(16) unsigned short As[128][40];  // [m][k], 64B data + 16B pad
  __shared__ __align__(16) unsigned short Bs[128][40];  // [n][k] transposed

  const int m0 = blockIdx.y * 128;
  const int n0 = blockIdx.x * 128;
  const int tid = threadIdx.x;
  const int wave = tid >> 5, lane = tid & 31;
  const int grp = lane >> 4, ln = lane & 15;
  const int wm = wave & 3;   // 4 M-groups of 32
  const int wn = wave >> 2;  // 2 N-groups of 64

  v8f acc[2][4];
#pragma unroll
  for (int i = 0; i < 2; ++i)
#pragma unroll
    for (int j = 0; j < 4; ++j)
#pragma unroll
      for (int e = 0; e < 8; ++e) acc[i][j][e] = 0.f;

  const unsigned asAddr = lds_addr_of(&As[0][0]);

  for (int k0 = 0; k0 < K; k0 += 32) {
    // ---- A tile via Tensor Data Mover: 128 rows x 32 bf16, row stride K,
    //      LDS pad 16B per 64B row (interval code 3 = 16 DW, amount code 3 = 4 DW)
    if (wave == 0) {
      tdm_load_2d_bf16(asAddr, A + (size_t)m0 * K + k0,
                       (unsigned)K, 8192u, (unsigned)K, 32u, 128u, 3u, 3u);
    }
    // ---- stage B tile transposed -> Bs[n][k]
    {
      int kr = tid >> 3, nc = (tid & 7) * 16;
      const uint4* src = (const uint4*)(B + (size_t)(k0 + kr) * N + n0 + nc);
      union { uint4 q; unsigned short s[8]; } u0, u1;
      u0.q = src[0]; u1.q = src[1];
#pragma unroll
      for (int j = 0; j < 8; ++j) Bs[nc + j][kr] = u0.s[j];
#pragma unroll
      for (int j = 0; j < 8; ++j) Bs[nc + 8 + j][kr] = u1.s[j];
      if (k0 + 32 < K)
        __builtin_prefetch((const void*)(B + (size_t)(k0 + 32 + kr) * N + n0 + nc), 0, 1);
    }
    if (wave == 0) __builtin_amdgcn_s_wait_tensorcnt(0);
    __syncthreads();

    Frag af[2];
#pragma unroll
    for (int mi = 0; mi < 2; ++mi) {
      const uint4* r = (const uint4*)&As[wm * 32 + mi * 16 + ln][0];
      af[mi].q[0] = r[grp];       // K = 8g .. 8g+7
      af[mi].q[1] = r[2 + grp];   // K = 16+8g .. 23+8g
    }
    Frag bfr[4];
#pragma unroll
    for (int ni = 0; ni < 4; ++ni) {
      const uint4* r = (const uint4*)&Bs[wn * 64 + ni * 16 + ln][0];
      bfr[ni].q[0] = r[grp];       // K = 8g .. 8g+7
      bfr[ni].q[1] = r[2 + grp];   // K = 16+8g .. 23+8g
    }
#pragma unroll
    for (int mi = 0; mi < 2; ++mi)
#pragma unroll
      for (int ni = 0; ni < 4; ++ni)
        acc[mi][ni] = wmma_bf(af[mi], bfr[ni], acc[mi][ni]);
    __syncthreads();
  }

  // ---- epilogue
#pragma unroll
  for (int mi = 0; mi < 2; ++mi)
#pragma unroll
    for (int ni = 0; ni < 4; ++ni) {
      int n = n0 + wn * 64 + ni * 16 + ln;
      float bb = bias[n];
#pragma unroll
      for (int r = 0; r < 8; ++r) {
        int m = m0 + wm * 32 + mi * 16 + r + grp * 8;
        float v = acc[mi][ni][r] + bb;
        if (mode == 1) {
          Out[(size_t)m * N + n] = v;
        } else {
          int t = n / CC, cc = n % CC;
          int head = cc >> 6, d = cc & 63;
          int b = m >> 10, pos = m & 1023;
          size_t idx = ((size_t)(b * NH + head) * HW + pos) * HD + d;
          if (t == 0)       qb[idx] = f2bf(v);
          else if (t == 1)  kb[idx] = f2bf(v * 0.125f);  // fold softmax scale into K
          else              vb[idx] = f2bf(v);
        }
      }
    }
}

// ---------------------------------------------------------------------------
// rel_h / rel_w bias tables:  out[bh, qrow, k] = sum_d q[bh,qrow,d]*table[t-k+31,d]
// ---------------------------------------------------------------------------
__global__ __launch_bounds__(256) void rel_bias(
    const unsigned short* __restrict__ qb, const float* __restrict__ table,
    float* __restrict__ out, int useH) {
  size_t gid = (size_t)blockIdx.x * 256 + threadIdx.x;  // < 96*1024*32
  int k = (int)(gid & 31);
  size_t qidx = gid >> 5;           // bh*1024 + qrow
  int qrow = (int)(qidx & 1023);
  int t = useH ? (qrow >> 5) : (qrow & 31);
  const unsigned short* qp = qb + qidx * HD;
  const float* rp = table + (size_t)(t - k + 31) * HD;
  float s = 0.f;
#pragma unroll
  for (int d = 0; d < HD; d += 4) {
    uint2 qu = *(const uint2*)(qp + d);
    float4 rv = *(const float4*)(rp + d);
    s += bf2f((unsigned short)(qu.x & 0xffff)) * rv.x;
    s += bf2f((unsigned short)(qu.x >> 16))    * rv.y;
    s += bf2f((unsigned short)(qu.y & 0xffff)) * rv.z;
    s += bf2f((unsigned short)(qu.y >> 16))    * rv.w;
  }
  out[gid] = s;
}

// ---------------------------------------------------------------------------
// Flash attention: one block = (head, 128 q-rows); wave owns 16 q rows.
// K chunk staged by TDM; V chunk transposed manually. Online softmax in regs.
// ---------------------------------------------------------------------------
__global__ __launch_bounds__(256) void flash_attn(
    const unsigned short* __restrict__ qb, const unsigned short* __restrict__ kb,
    const unsigned short* __restrict__ vb, const float* __restrict__ relh,
    const float* __restrict__ relw, unsigned short* __restrict__ ob) {
  __shared__ __align__(16) unsigned short Ks[32][72];      // [key][hd], 128B+16B pad
  __shared__ __align__(16) unsigned short Vs[64][40];      // [hd][key] transposed
  __shared__ __align__(16) unsigned short Ps[8][16][40];   // per-wave P staging

  const int bh = blockIdx.x >> 3;
  const int qblk = blockIdx.x & 7;
  const int tid = threadIdx.x, wave = tid >> 5, lane = tid & 31;
  const int grp = lane >> 4, ln = lane & 15;
  const int qbase = qblk * 128 + wave * 16;     // q row within head
  const size_t headBase = (size_t)bh * HW;

  // Q A-fragments for hd 0..31 and 32..63 (per-lane row = ln)
  Frag qa[2];
  {
    const uint4* qp = (const uint4*)(qb + (headBase + qbase + ln) * HD);
    qa[0].q[0] = qp[grp];     qa[0].q[1] = qp[2 + grp];
    qa[1].q[0] = qp[4 + grp]; qa[1].q[1] = qp[6 + grp];
  }

  float mrow[8], lrow[8];
  v8f oacc[4];
#pragma unroll
  for (int r = 0; r < 8; ++r) { mrow[r] = -1e30f; lrow[r] = 0.f; }
#pragma unroll
  for (int c = 0; c < 4; ++c)
#pragma unroll
    for (int e = 0; e < 8; ++e) oacc[c][e] = 0.f;

  const unsigned ksAddr = lds_addr_of(&Ks[0][0]);

  for (int kc = 0; kc < HW; kc += 32) {
    // ---- K chunk via TDM: 32 rows x 64 bf16, row stride 64,
    //      pad 16B per 128B row (interval code 4 = 32 DW, amount code 3 = 4 DW)
    if (wave == 0) {
      tdm_load_2d_bf16(ksAddr, kb + (headBase + kc) * HD,
                       (unsigned)HD, (unsigned)(BH * HW), (unsigned)HD,
                       (unsigned)HD, 32u, 4u, 3u);
    }
    // ---- stage V chunk transposed
    {
      int key = tid >> 3, hdo = (tid & 7) * 8;
      uint4 vv = *(const uint4*)(vb + (headBase + kc + key) * HD + hdo);
      union { uint4 q; unsigned short s[8]; } u;
      u.q = vv;
#pragma unroll
      for (int j = 0; j < 8; ++j) Vs[hdo + j][key] = u.s[j];
    }
    if (wave == 0) __builtin_amdgcn_s_wait_tensorcnt(0);
    __syncthreads();

    // ---- scores: two 16x16 tiles (keys kt*16..kt*16+15), K-dim = hd (2 wmma)
    v8f S[2];
#pragma unroll
    for (int kt = 0; kt < 2; ++kt) {
      const uint4* rk = (const uint4*)&Ks[kt * 16 + ln][0];
      Frag bk0, bk1;
      bk0.q[0] = rk[grp];     bk0.q[1] = rk[2 + grp];     // hd 0..31
      bk1.q[0] = rk[4 + grp]; bk1.q[1] = rk[6 + grp]; // hd 32..63
      v8f s;
#pragma unroll
      for (int e = 0; e < 8; ++e) s[e] = 0.f;
      s = wmma_bf(qa[0], bk0, s);
      s = wmma_bf(qa[1], bk1, s);
      // decomposed rel-pos bias
      int key = kc + kt * 16 + ln;
      int kh = key >> 5, kw = key & 31;
#pragma unroll
      for (int r = 0; r < 8; ++r) {
        size_t q32 = (headBase + qbase + r + grp * 8) * 32;
        s[r] += relh[q32 + kh] + relw[q32 + kw];
      }
      S[kt] = s;
    }

    // ---- online softmax (row stats shared across each 16-lane group)
#pragma unroll
    for (int r = 0; r < 8; ++r) {
      float mx = fmaxf(S[0][r], S[1][r]);
#pragma unroll
      for (int off = 8; off > 0; off >>= 1) mx = fmaxf(mx, __shfl_xor(mx, off, 32));
      float mnew = fmaxf(mrow[r], mx);
      float sc = __expf(mrow[r] - mnew);
      mrow[r] = mnew;
      float p0 = __expf(S[0][r] - mnew);
      float p1 = __expf(S[1][r] - mnew);
      float rs = p0 + p1;
#pragma unroll
      for (int off = 8; off > 0; off >>= 1) rs += __shfl_xor(rs, off, 32);
      lrow[r] = lrow[r] * sc + rs;
#pragma unroll
      for (int c = 0; c < 4; ++c) oacc[c][r] *= sc;
      S[0][r] = p0; S[1][r] = p1;
    }

    // ---- P (C-layout) -> LDS -> A-fragment layout
#pragma unroll
    for (int r = 0; r < 8; ++r) {
      int m = r + grp * 8;
      Ps[wave][m][ln]      = f2bf(S[0][r]);
      Ps[wave][m][16 + ln] = f2bf(S[1][r]);
    }
    asm volatile("s_wait_dscnt 0" ::: "memory");

    Frag pa;
    {
      const uint4* rp = (const uint4*)&Ps[wave][ln][0];
      pa.q[0] = rp[grp]; pa.q[1] = rp[2 + grp];
    }
#pragma unroll
    for (int c = 0; c < 4; ++c) {
      const uint4* rv = (const uint4*)&Vs[c * 16 + ln][0];
      Frag bv;
      bv.q[0] = rv[grp]; bv.q[1] = rv[2 + grp];
      oacc[c] = wmma_bf(pa, bv, oacc[c]);
    }
    __syncthreads();
  }

  // ---- normalize + store to (row, head*64+d) bf16
  const int b = bh / NH, head = bh % NH;
#pragma unroll
  for (int c = 0; c < 4; ++c)
#pragma unroll
    for (int r = 0; r < 8; ++r) {
      int row = b * HW + qbase + r + grp * 8;
      int col = head * HD + c * 16 + ln;
      ob[(size_t)row * CC + col] = f2bf(oacc[c][r] / lrow[r]);
    }
}

// ---------------------------------------------------------------------------
// Launch
// ---------------------------------------------------------------------------
extern "C" void kernel_launch(void* const* d_in, const int* in_sizes, int n_in,
                              void* d_out, int out_size, void* d_ws, size_t ws_size,
                              hipStream_t stream) {
  const float* x      = (const float*)d_in[0];
  const float* w_qkv  = (const float*)d_in[1];
  const float* b_qkv  = (const float*)d_in[2];
  const float* w_proj = (const float*)d_in[3];
  const float* b_proj = (const float*)d_in[4];
  const float* rp_h   = (const float*)d_in[5];
  const float* rp_w   = (const float*)d_in[6];
  float* out = (float*)d_out;
  char* ws = (char*)d_ws;

  // workspace layout (bytes)
  const size_t SZ_XBF   = (size_t)ROWS * CC * 2;        // 12.6 MB
  const size_t SZ_WQKV  = (size_t)CC * 3 * CC * 2;      //  3.5 MB
  const size_t SZ_WPROJ = (size_t)CC * CC * 2;          //  1.2 MB
  const size_t SZ_QKV   = (size_t)BH * HW * HD * 2;     // 12.6 MB each
  const size_t SZ_REL   = (size_t)BH * HW * 32 * 4;     // 12.6 MB each
  size_t off = 0;
  unsigned short* x_bf     = (unsigned short*)(ws + off); off += SZ_XBF;
  unsigned short* wqkv_bf  = (unsigned short*)(ws + off); off += SZ_WQKV;
  unsigned short* wproj_bf = (unsigned short*)(ws + off); off += SZ_WPROJ;
  unsigned short* qbf      = (unsigned short*)(ws + off); off += SZ_QKV;
  unsigned short* kbf      = (unsigned short*)(ws + off); off += SZ_QKV;
  unsigned short* vbf      = (unsigned short*)(ws + off); off += SZ_QKV;
  float* relh              = (float*)(ws + off);          off += SZ_REL;
  float* relw              = (float*)(ws + off);          off += SZ_REL;
  unsigned short* obf      = (unsigned short*)(ws + off); off += SZ_XBF;
  (void)in_sizes; (void)n_in; (void)out_size; (void)ws_size;

  // 1) fp32 -> bf16
  cvt_bf16<<<(ROWS * CC) / 1024, 256, 0, stream>>>(x, x_bf, ROWS * CC);
  cvt_bf16<<<(CC * 3 * CC) / 1024, 256, 0, stream>>>(w_qkv, wqkv_bf, CC * 3 * CC);
  cvt_bf16<<<(CC * CC) / 1024, 256, 0, stream>>>(w_proj, wproj_bf, CC * CC);

  // 2) QKV GEMM (8192 x 768) @ (768 x 2304), scatter into per-head q/k/v
  gemm_bf16<<<dim3((3 * CC) / 128, ROWS / 128), 256, 0, stream>>>(
      x_bf, wqkv_bf, b_qkv, CC, 3 * CC, 0, nullptr, qbf, kbf, vbf);

  // 3) decomposed rel-pos bias tables
  rel_bias<<<(BH * HW * 32) / 256, 256, 0, stream>>>(qbf, rp_h, relh, 1);
  rel_bias<<<(BH * HW * 32) / 256, 256, 0, stream>>>(qbf, rp_w, relw, 0);

  // 4) flash attention (96 heads x 8 q-blocks)
  flash_attn<<<BH * 8, 256, 0, stream>>>(qbf, kbf, vbf, relh, relw, obf);

  // 5) output projection (8192 x 768) @ (768 x 768) + bias -> fp32
  gemm_bf16<<<dim3(CC / 128, ROWS / 128), 256, 0, stream>>>(
      obf, wproj_bf, b_proj, CC, CC, 1, out, nullptr, nullptr, nullptr);
}
